// RNN_48344151884388
// MI455X (gfx1250) — compile-verified, matrix operand order checked by reading, not run on hardware
//
#include <hip/hip_runtime.h>
#include <cstddef>

typedef __attribute__((ext_vector_type(16))) __bf16 v16bf;
typedef __attribute__((ext_vector_type(8)))  __bf16 v8bf;
typedef __attribute__((ext_vector_type(8)))  float  v8f;

constexpr int B_ = 256, T_ = 512, I_ = 300, H_ = 128;
constexpr int KP  = 320;        // I padded to multiple of 32
constexpr int LDX = KP + 8;     // LDS stride (elements), bank-conflict free
constexpr int LDW = KP + 8;
constexpr int LDH = H_ + 8;
constexpr int MT1 = 64;         // BT rows per workgroup in phase 1

// ---------------- Phase 1: x_proj[bt,h] = sum_i x[bt,i]*W_ih[h,i] + b_ih[h] + b_hh[h]
__global__ void __launch_bounds__(256) xproj_kernel(
    const float* __restrict__ x, const float* __restrict__ W_ih,
    const float* __restrict__ b_ih, const float* __restrict__ b_hh,
    float* __restrict__ Y) {
  extern __shared__ char smem[];
  __bf16* sW = (__bf16*)smem;            // [H_][LDW]
  __bf16* sX = sW + H_ * LDW;            // [MT1][LDX]

  const int tid = threadIdx.x;
  const int bt0 = blockIdx.x * MT1;

  // Stage W_ih (128x300 f32) -> bf16 LDS, zero-pad K 300..319
  for (int idx = tid; idx < H_ * KP; idx += 256) {
    int r = idx / KP, c = idx - r * KP;
    float v = (c < I_) ? W_ih[r * I_ + c] : 0.f;
    sW[r * LDW + c] = (__bf16)v;
  }
  // Stage x tile (64x300 f32) -> bf16 LDS, zero-pad
  for (int idx = tid; idx < MT1 * KP; idx += 256) {
    int r = idx / KP, c = idx - r * KP;
    float v = (c < I_) ? x[(size_t)(bt0 + r) * I_ + c] : 0.f;
    sX[r * LDX + c] = (__bf16)v;
  }
  __syncthreads();

  const int lane = tid & 31;
  const int wv   = tid >> 5;      // 8 waves
  const int mt   = wv & 3;        // M tile 0..3 (16 rows each)
  const int ng   = wv >> 2;       // N group: tiles ng*4 .. ng*4+3
  const int lm   = lane & 15;
  const int kh   = (lane >> 4) * 8;

  v8f acc[4] = {};
  const __bf16* aRow = sX + (mt * 16 + lm) * LDX;
  for (int kc = 0; kc < KP; kc += 32) {
    v16bf af;
    {
      v8bf lo = *(const v8bf*)(aRow + kc + kh);
      v8bf hi = *(const v8bf*)(aRow + kc + kh + 16);
      #pragma unroll
      for (int i = 0; i < 8; ++i) { af[i] = lo[i]; af[8 + i] = hi[i]; }
    }
    #pragma unroll
    for (int j = 0; j < 4; ++j) {
      const __bf16* bRow = sW + ((ng * 4 + j) * 16 + lm) * LDW;
      v16bf bfr;
      v8bf lo = *(const v8bf*)(bRow + kc + kh);
      v8bf hi = *(const v8bf*)(bRow + kc + kh + 16);
      #pragma unroll
      for (int i = 0; i < 8; ++i) { bfr[i] = lo[i]; bfr[8 + i] = hi[i]; }
      acc[j] = __builtin_amdgcn_wmma_f32_16x16x32_bf16(
          false, af, false, bfr, (short)0, acc[j], false, false);
    }
  }

  const int rbase = (lane >> 4) * 8;           // C layout rows
  #pragma unroll
  for (int j = 0; j < 4; ++j) {
    int col = (ng * 4 + j) * 16 + lm;
    float bias = b_ih[col] + b_hh[col];
    #pragma unroll
    for (int r = 0; r < 8; ++r) {
      int row = bt0 + mt * 16 + rbase + r;
      Y[(size_t)row * H_ + col] = acc[j][r] + bias;
    }
  }
}

// ---------------- Phase 2: serial scan  h = tanh(Y_t + h @ W_hh^T)
__global__ void __launch_bounds__(256) rnn_scan_kernel(
    const float* __restrict__ Y, const float* __restrict__ W_hh,
    float* __restrict__ hlast) {
  extern __shared__ char smem[];
  __bf16* sWhh = (__bf16*)smem;            // [128][LDH]
  __bf16* sH   = sWhh + H_ * LDH;          // double-buffered [2][16][LDH]

  const int tid = threadIdx.x;
  const int b0  = blockIdx.x * 16;

  for (int idx = tid; idx < H_ * H_; idx += 256) {
    int r = idx >> 7, c = idx & 127;
    sWhh[r * LDH + c] = (__bf16)W_hh[idx];
  }
  for (int idx = tid; idx < 2 * 16 * LDH; idx += 256)
    sH[idx] = (__bf16)0.f;                 // h0 = 0 (both buffers)
  __syncthreads();

  const int lane  = tid & 31;
  const int wv    = tid >> 5;              // N tile 0..7
  const int lm    = lane & 15;
  const int kh    = (lane >> 4) * 8;
  const int n0    = wv * 16;
  const int col   = n0 + lm;
  const int rbase = (lane >> 4) * 8;

  // Loop-invariant W_hh B-fragments (B[k,n] = W_hh[n,k]) — 4 chunks = 32 VGPRs
  v16bf wfrag[4];
  {
    const __bf16* bRow = sWhh + (n0 + lm) * LDH;
    #pragma unroll
    for (int kc = 0; kc < 4; ++kc) {
      v8bf lo = *(const v8bf*)(bRow + kc * 32 + kh);
      v8bf hi = *(const v8bf*)(bRow + kc * 32 + kh + 16);
      #pragma unroll
      for (int i = 0; i < 8; ++i) { wfrag[kc][i] = lo[i]; wfrag[kc][8 + i] = hi[i]; }
    }
  }

  // Prefetch x_proj for t=0 (L2-resident)
  float ynext[8];
  #pragma unroll
  for (int r = 0; r < 8; ++r)
    ynext[r] = Y[(size_t)((b0 + rbase + r) * T_ + 0) * H_ + col];

  float th[8];
  int p = 0;
  for (int t = 0; t < T_; ++t) {
    float ycur[8];
    #pragma unroll
    for (int r = 0; r < 8; ++r) ycur[r] = ynext[r];
    if (t + 1 < T_) {                      // prefetch next timestep during compute
      #pragma unroll
      for (int r = 0; r < 8; ++r)
        ynext[r] = Y[(size_t)((b0 + rbase + r) * T_ + (t + 1)) * H_ + col];
    }

    const __bf16* aRow = sH + p * 16 * LDH + lm * LDH;
    v8f acc = {};
    #pragma unroll
    for (int kc = 0; kc < 4; ++kc) {
      v16bf af;
      v8bf lo = *(const v8bf*)(aRow + kc * 32 + kh);
      v8bf hi = *(const v8bf*)(aRow + kc * 32 + kh + 16);
      #pragma unroll
      for (int i = 0; i < 8; ++i) { af[i] = lo[i]; af[8 + i] = hi[i]; }
      acc = __builtin_amdgcn_wmma_f32_16x16x32_bf16(
          false, af, false, wfrag[kc], (short)0, acc, false, false);
    }

    #pragma unroll
    for (int r = 0; r < 8; ++r) th[r] = tanhf(acc[r] + ycur[r]);

    __bf16* hw = sH + (1 - p) * 16 * LDH;  // write to other buffer
    #pragma unroll
    for (int r = 0; r < 8; ++r) hw[(rbase + r) * LDH + col] = (__bf16)th[r];
    p ^= 1;
    __syncthreads();                       // single barrier per step
  }

  #pragma unroll
  for (int r = 0; r < 8; ++r)
    hlast[(size_t)(b0 + rbase + r) * H_ + col] = th[r];
}

// ---------------- Phase 3: tiny fc head  out = h_last @ W_fc^T + b_fc
__global__ void __launch_bounds__(256) fc_kernel(
    const float* __restrict__ hlast, const float* __restrict__ W_fc,
    const float* __restrict__ b_fc, float* __restrict__ out) {
  int b = blockIdx.x * blockDim.x + threadIdx.x;
  if (b >= B_) return;
  float a0 = b_fc[0], a1 = b_fc[1];
  for (int h = 0; h < H_; ++h) {
    float hv = hlast[b * H_ + h];
    a0 += hv * W_fc[h];
    a1 += hv * W_fc[H_ + h];
  }
  out[b * 2 + 0] = a0;
  out[b * 2 + 1] = a1;
}

extern "C" void kernel_launch(void* const* d_in, const int* in_sizes, int n_in,
                              void* d_out, int out_size, void* d_ws, size_t ws_size,
                              hipStream_t stream) {
  const float* x    = (const float*)d_in[0];
  const float* W_ih = (const float*)d_in[1];
  const float* W_hh = (const float*)d_in[2];
  const float* b_ih = (const float*)d_in[3];
  const float* b_hh = (const float*)d_in[4];
  const float* W_fc = (const float*)d_in[5];
  const float* b_fc = (const float*)d_in[6];
  float* out = (float*)d_out;

  float* Y     = (float*)d_ws;                  // [B*T, H] = 64 MB (L2-resident)
  float* hlast = Y + (size_t)B_ * T_ * H_;      // [B, H]

  size_t lds1 = (size_t)(H_ * LDW + MT1 * LDX) * sizeof(__bf16);
  xproj_kernel<<<(B_ * T_) / MT1, 256, lds1, stream>>>(x, W_ih, b_ih, b_hh, Y);

  size_t lds2 = (size_t)(H_ * LDH + 2 * 16 * LDH) * sizeof(__bf16);
  rnn_scan_kernel<<<B_ / 16, 256, lds2, stream>>>(Y, W_hh, hlast);

  fc_kernel<<<1, 256, 0, stream>>>(hlast, W_fc, b_fc, out);
}
